// RandomProjectionModule_16836271800994
// MI455X (gfx1250) — matrix operand order repaired and checked by reading, not compiled, not run on hardware
//
#include <hip/hip_runtime.h>
#include <hip/hip_bf16.h>
#include <math.h>
#include <stdint.h>

// Problem constants (from reference)
#define N_NODES 100000
#define DIM     256
#define NEDGE   200000
#define NPAIR   50000
#define WDK     1e-6f
#define FDIM    64     // pairwise feature dim (8x8 gram)
#define HDIM    256    // MLP hidden dim
#define EPB     4      // edges per block in scatter kernel

typedef __attribute__((ext_vector_type(16))) _Float16 v16h;
typedef __attribute__((ext_vector_type(8)))  float    v8f;
typedef int b128i __attribute__((vector_size(16)));   // 4-DWORD payload type

// ---------------------------------------------------------------------------
// CDNA5 async global->LDS copy (ASYNCcnt-tracked), with safe fallback.
// Builtin signature (from hipcc diagnostic): param0 = AS(1) v4i* global src.
// ---------------------------------------------------------------------------
#if defined(__gfx1250__) && __has_builtin(__builtin_amdgcn_global_load_async_to_lds_b128)
#define HAVE_ASYNC_B128 1
#else
#define HAVE_ASYNC_B128 0
#endif

__device__ __forceinline__ void lds_copy16_async(float* l, const float* g) {
#if HAVE_ASYNC_B128
    __builtin_amdgcn_global_load_async_to_lds_b128(
        (__attribute__((address_space(1))) b128i*)(uintptr_t)g,
        (__attribute__((address_space(3))) b128i*)(uintptr_t)l,
        /*offset=*/0, /*cpol=*/0);
#else
    *(float4*)l = *(const float4*)g;
#endif
}

__device__ __forceinline__ void async_wait_all() {
#if HAVE_ASYNC_B128
#if __has_builtin(__builtin_amdgcn_s_wait_asynccnt)
    __builtin_amdgcn_s_wait_asynccnt(0);
#else
    asm volatile("s_wait_asynccnt 0x0" ::: "memory");
#endif
#endif
}

// ---------------------------------------------------------------------------
// Kernel 1: initialize accumulated tables  new_i = rp_i * decay^i  (i = 1..3)
// ---------------------------------------------------------------------------
__global__ void k_init_tables(const float* __restrict__ rp1,
                              const float* __restrict__ rp2,
                              const float* __restrict__ rp3,
                              const float* __restrict__ times,
                              const float* __restrict__ now_time,
                              float* __restrict__ ws) {
    const long long per4 = (long long)N_NODES * DIM / 4;      // float4 per table
    long long idx = (long long)blockIdx.x * blockDim.x + threadIdx.x;
    if (idx >= 3 * per4) return;
    float next_t = times[NEDGE - 1];
    float decay  = __expf(-WDK * (next_t - now_time[0]));
    int table = (int)(idx / per4);
    long long off = idx - (long long)table * per4;
    const float4* src = (table == 0) ? (const float4*)rp1
                      : (table == 1) ? (const float4*)rp2
                                     : (const float4*)rp3;
    float s = (table == 0) ? decay : (table == 1) ? decay * decay
                                                  : decay * decay * decay;
    float4 v = src[off];
    v.x *= s; v.y *= s; v.z *= s; v.w *= s;
    ((float4*)ws)[(long long)table * per4 + off] = v;
}

// ---------------------------------------------------------------------------
// Kernel 2: edge scatter. One block (256 threads) handles EPB edges; prefetch
// the next edge's six source rows (global_prefetch_b8) while the current
// edge's gathers + f32 atomics are in flight. All layers read the pre-update
// decayed base tables, so a single atomic pass is exact.
// ---------------------------------------------------------------------------
__global__ void __launch_bounds__(DIM)
k_scatter(const float* __restrict__ rp0, const float* __restrict__ rp1,
          const float* __restrict__ rp2,
          const float* __restrict__ times, const float* __restrict__ now_time,
          const int* __restrict__ src_ids, const int* __restrict__ dst_ids,
          float* __restrict__ new1, float* __restrict__ new2,
          float* __restrict__ new3) {
    int d  = threadIdx.x;
    int e0 = blockIdx.x * EPB;
    float next_t = times[NEDGE - 1];
    float decay  = __expf(-WDK * (next_t - now_time[0]));
    float dd2    = decay * decay;

    int   ss[EPB], dd[EPB];
    float tws[EPB];
    #pragma unroll
    for (int j = 0; j < EPB; ++j) {
        int e  = e0 + j;
        ss[j]  = src_ids[e];
        dd[j]  = dst_ids[e];
        tws[j] = __expf(-WDK * (next_t - times[e]));
    }

    #pragma unroll
    for (int j = 0; j < EPB; ++j) {
        if (j + 1 < EPB) {   // prefetch next edge's source rows
            long long pso = (long long)ss[j + 1] * DIM + d;
            long long pto = (long long)dd[j + 1] * DIM + d;
            __builtin_prefetch(&rp0[pso], 0, 3);
            __builtin_prefetch(&rp0[pto], 0, 3);
            __builtin_prefetch(&rp1[pso], 0, 3);
            __builtin_prefetch(&rp1[pto], 0, 3);
            __builtin_prefetch(&rp2[pso], 0, 3);
            __builtin_prefetch(&rp2[pto], 0, 3);
        }
        long long so = (long long)ss[j] * DIM + d;
        long long to = (long long)dd[j] * DIM + d;
        float tw = tws[j];
        float c1 = decay * tw;
        float c2 = dd2 * tw;
        float v0t = rp0[to], v0s = rp0[so];
        float v1t = rp1[to], v1s = rp1[so];
        float v2t = rp2[to], v2s = rp2[so];
        atomicAdd(&new1[so], v0t * tw);
        atomicAdd(&new1[to], v0s * tw);
        atomicAdd(&new2[so], v1t * c1);
        atomicAdd(&new2[to], v1s * c1);
        atomicAdd(&new3[so], v2t * c2);
        atomicAdd(&new3[to], v2s * c2);
    }
}

// ---------------------------------------------------------------------------
// Kernel 3: per-pair 8x256 projection gather -> 8x8 gram -> log1p(relu).
// One block of 64 threads per pair. Rows are streamed straight into LDS with
// GLOBAL_LOAD_ASYNC_TO_LDS_B128 (ASYNCcnt), then thread t computes gram entry
// (p,q) = (t/8, t%8) from LDS with float4 dot products.
// ---------------------------------------------------------------------------
__global__ void __launch_bounds__(64)
k_feat(const float* __restrict__ rp0, const float* __restrict__ new1,
       const float* __restrict__ new2, const float* __restrict__ new3,
       const int* __restrict__ q_src, const int* __restrict__ q_dst,
       float* __restrict__ feat) {
    __shared__ float rows[8 * DIM];
    int b = blockIdx.x;
    int t = threadIdx.x;
    int sn = q_src[b], dn = q_dst[b];
    for (int i = t; i < 8 * (DIM / 4); i += 64) {
        int r  = i >> 6;          // row 0..7
        int c4 = i & 63;          // float4 index within row
        int node = (r < 4) ? sn : dn;
        int lay  = r & 3;
        const float* tab = (lay == 0) ? rp0 : (lay == 1) ? new1
                         : (lay == 2) ? new2 : new3;
        lds_copy16_async(rows + r * DIM + c4 * 4,
                         tab + (long long)node * DIM + c4 * 4);
    }
    async_wait_all();
    __syncthreads();
    int p = t >> 3, q = t & 7;
    const float4* rpv = (const float4*)(rows + p * DIM);
    const float4* rqv = (const float4*)(rows + q * DIM);
    float acc = 0.f;
    #pragma unroll 4
    for (int i = 0; i < DIM / 4; ++i) {
        float4 a = rpv[i], c = rqv[i];
        acc = fmaf(a.x, c.x, acc);
        acc = fmaf(a.y, c.y, acc);
        acc = fmaf(a.z, c.z, acc);
        acc = fmaf(a.w, c.w, acc);
    }
    acc = fmaxf(acc, 0.f);
    feat[(long long)b * FDIM + t] = log1pf(acc);
}

// ---------------------------------------------------------------------------
// Kernel 4: fused MLP  out = relu(feat@w1 + b1) @ w2 + b2  via WMMA f16->f32.
// One wave per 16 batch rows. ISA 16-bit A/B fragment swizzles per cdna5 7.12.2.
// ---------------------------------------------------------------------------
__device__ __forceinline__ int a_k_of(int lane, int i) {
    // A (MxK, 16x32, f16): lane<16 -> K {0..7,16..23}; lane>=16 -> {8..15,24..31}
    return (i & 7) + 8 * (((i >> 3) << 1) + (lane >> 4));
}

__global__ void __launch_bounds__(32)
k_mlp(const float* __restrict__ feat,
      const float* __restrict__ w1, const float* __restrict__ b1,
      const float* __restrict__ w2, const float* __restrict__ b2,
      float* __restrict__ out) {
    __shared__ _Float16 lds_a[16 * FDIM];   // feat tile   (16 x 64)  f16
    __shared__ _Float16 lds_h[16 * HDIM];   // hidden tile (16 x 256) f16
    int lane = threadIdx.x;
    long long rowbase = (long long)blockIdx.x * 16;

    // Stage feat tile to LDS as f16
    for (int i = lane; i < 16 * FDIM; i += 32)
        lds_a[i] = (_Float16)feat[(rowbase + (i >> 6)) * FDIM + (i & 63)];
    __syncthreads();

    const int nlo   = lane & 15;
    const int khalf = lane >> 4;

    // GEMM1: [16x64] @ [64x256], N in 16 tiles of 16, K in 2 chunks of 32
    for (int nt = 0; nt < 16; ++nt) {
        int n = nt * 16 + nlo;
        v8f acc = {};
        for (int kc = 0; kc < 2; ++kc) {
            int kbase = kc * 32;
            v16h a, bf;
            #pragma unroll
            for (int i = 0; i < 16; ++i)
                a[i] = lds_a[nlo * FDIM + kbase + a_k_of(lane, i)];
            #pragma unroll
            for (int i = 0; i < 16; ++i)
                bf[i] = (_Float16)w1[(kbase + a_k_of(lane, i)) * HDIM + n];
            acc = __builtin_amdgcn_wmma_f32_16x16x32_f16(
                false, a, false, bf, (short)0, acc, false, false);
        }
        float bias = b1[n];
        #pragma unroll
        for (int r = 0; r < 8; ++r) {
            int m = r + 8 * khalf;
            lds_h[m * HDIM + n] = (_Float16)fmaxf(acc[r] + bias, 0.f);
        }
    }
    __syncthreads();

    // GEMM2: [16x256] @ [256x64], N in 4 tiles of 16, K in 8 chunks of 32
    for (int nt = 0; nt < 4; ++nt) {
        int n = nt * 16 + nlo;
        v8f acc = {};
        for (int kc = 0; kc < 8; ++kc) {
            int kbase = kc * 32;
            v16h a, bf;
            #pragma unroll
            for (int i = 0; i < 16; ++i)
                a[i] = lds_h[nlo * HDIM + kbase + a_k_of(lane, i)];
            #pragma unroll
            for (int i = 0; i < 16; ++i)
                bf[i] = (_Float16)w2[(kbase + a_k_of(lane, i)) * FDIM + n];
            acc = __builtin_amdgcn_wmma_f32_16x16x32_f16(
                false, a, false, bf, (short)0, acc, false, false);
        }
        float bias = b2[n];
        #pragma unroll
        for (int r = 0; r < 8; ++r) {
            int m = r + 8 * khalf;
            out[(rowbase + m) * FDIM + n] = acc[r] + bias;
        }
    }
}

// ---------------------------------------------------------------------------
extern "C" void kernel_launch(void* const* d_in, const int* in_sizes, int n_in,
                              void* d_out, int out_size, void* d_ws, size_t ws_size,
                              hipStream_t stream) {
    const float* rp0      = (const float*)d_in[0];
    const float* rp1      = (const float*)d_in[1];
    const float* rp2      = (const float*)d_in[2];
    const float* rp3      = (const float*)d_in[3];
    const float* times    = (const float*)d_in[4];
    const float* now_time = (const float*)d_in[5];
    const float* w1       = (const float*)d_in[6];
    const float* b1       = (const float*)d_in[7];
    const float* w2       = (const float*)d_in[8];
    const float* b2       = (const float*)d_in[9];
    const int*   src_ids  = (const int*)d_in[10];
    const int*   dst_ids  = (const int*)d_in[11];
    const int*   q_src    = (const int*)d_in[12];
    const int*   q_dst    = (const int*)d_in[13];

    float* ws   = (float*)d_ws;
    const size_t tab = (size_t)N_NODES * DIM;
    float* new1 = ws;
    float* new2 = ws + tab;
    float* new3 = ws + 2 * tab;
    float* feat = ws + 3 * tab;
    float* outp = (float*)d_out;

    {   // init: 3 * N * D / 4 float4 elements
        long long tot4 = (long long)3 * N_NODES * DIM / 4;
        int blocks = (int)((tot4 + 255) / 256);
        k_init_tables<<<blocks, 256, 0, stream>>>(rp1, rp2, rp3, times, now_time, ws);
    }
    k_scatter<<<NEDGE / EPB, DIM, 0, stream>>>(rp0, rp1, rp2, times, now_time,
                                               src_ids, dst_ids, new1, new2, new3);
    k_feat<<<NPAIR, 64, 0, stream>>>(rp0, new1, new2, new3, q_src, q_dst, feat);
    k_mlp<<<NPAIR / 16, 32, 0, stream>>>(feat, w1, b1, w2, b2, outp);
}
